// LoraFluxAttnProcessor_1022202216576
// MI455X (gfx1250) — matrix operand order repaired and checked by reading, not compiled, not run on hardware
//
#include <hip/hip_runtime.h>
#include <cstdint>

typedef _Float16 half_t;
typedef __attribute__((ext_vector_type(16))) _Float16 v16h;
typedef __attribute__((ext_vector_type(8)))  float    v8f;

#define H_NUM   24
#define DH      128
#define SEQLEN  2048
#define DMODEL  3072
#define LORA_R  16

// Optional CDNA5 async global->LDS path (GLOBAL_LOAD_ASYNC_TO_LDS_B128, ASYNCcnt).
#if __has_builtin(__builtin_amdgcn_global_load_async_to_lds_b128) && \
    __has_builtin(__builtin_amdgcn_s_wait_asynccnt)
#define HAVE_ASYNC_LDS 1
#else
#define HAVE_ASYNC_LDS 0
#endif

#if HAVE_ASYNC_LDS
// Builtin signature (from hipcc diagnostic): param0 is
// '__attribute__((__vector_size__(4*sizeof(int)))) int __device__ *' (AS1 int4*).
typedef int v4i_vs __attribute__((vector_size(16)));
typedef __attribute__((address_space(1))) v4i_vs as1_v4i;
typedef __attribute__((address_space(3))) v4i_vs as3_v4i;
#endif

// ---------------------------------------------------------------------------
// WMMA helper: D = A(16x32 f16) * B(32x16 f16) + C(16x16 f32)
// ---------------------------------------------------------------------------
static __device__ __forceinline__ v8f wmma16x16x32(v16h a, v16h b, v8f c) {
    return __builtin_amdgcn_wmma_f32_16x16x32_f16(
        /*neg_a=*/false, a, /*neg_b=*/false, b,
        /*c_mod=*/(short)0, c, /*reuse_a=*/false, /*reuse_b=*/false);
}

// A-matrix fragment (16x32, row-major source, stride ld in halves).
// ISA 7.12.2: lane = h*16+m ; VGPR v holds K pair k=(v>>2)*16 + h*8 + (v&3)*2
static __device__ __forceinline__ v16h load_a_frag(const half_t* __restrict__ base, int ld) {
    int lane = threadIdx.x & 31;
    int m = lane & 15, h = lane >> 4;
    const half_t* __restrict__ row = base + (size_t)m * ld;
    v16h a;
#pragma unroll
    for (int v = 0; v < 8; ++v) {
        int k = ((v >> 2) << 4) + (h << 3) + ((v & 3) << 1);
        a[2 * v]     = row[k];
        a[2 * v + 1] = row[k + 1];
    }
    return a;
}

// B-matrix fragment (32x16): B[k][n] = W[n][k] with W row-major stride ld.
// lane = h*16+n ; VGPR v holds K pair k=(v>>2)*16 + h*8 + (v&3)*2
static __device__ __forceinline__ v16h load_b_frag(const half_t* __restrict__ base, int ld) {
    int lane = threadIdx.x & 31;
    int n = lane & 15, h = lane >> 4;
    const half_t* __restrict__ row = base + (size_t)n * ld;
    v16h b;
#pragma unroll
    for (int v = 0; v < 8; ++v) {
        int k = ((v >> 2) << 4) + (h << 3) + ((v & 3) << 1);
        b[2 * v]     = row[k];
        b[2 * v + 1] = row[k + 1];
    }
    return b;
}

// ---------------------------------------------------------------------------
// Kernel 1: fold LoRA into weight, downconvert: W_eff[n][k] = W + sum_r up*down
// ---------------------------------------------------------------------------
__global__ void fuse_lora_weight(const float* __restrict__ w,
                                 const float* __restrict__ up,    // [N][R]
                                 const float* __restrict__ down,  // [R][K]
                                 half_t* __restrict__ out, int N, int K) {
    size_t i = (size_t)blockIdx.x * blockDim.x + threadIdx.x;
    if (i >= (size_t)N * K) return;
    int n = (int)(i / K), k = (int)(i % K);
    float acc = w[i];
#pragma unroll
    for (int r = 0; r < LORA_R; ++r)
        acc += up[(size_t)n * LORA_R + r] * down[(size_t)r * K + k];
    out[i] = (half_t)acc;
}

// ---------------------------------------------------------------------------
// Kernel 2: fp32 -> fp16 elementwise
// ---------------------------------------------------------------------------
__global__ void cvt_f32_f16(const float* __restrict__ in, half_t* __restrict__ out, size_t n) {
    size_t i = (size_t)blockIdx.x * blockDim.x + threadIdx.x;
    if (i < n) out[i] = (half_t)in[i];
}

// ---------------------------------------------------------------------------
// Kernel 3/7: C[M][N] = A[M][K] * B[N][K]^T + bias ; 128x128 block, BK=32,
// 256 threads = 8 waves in a 4(M)x2(N) wave grid, each wave 2x4 WMMA tiles.
// ---------------------------------------------------------------------------
template <bool OUT_HALF>
__global__ void __launch_bounds__(256, 2)
gemm_nt_f16(const half_t* __restrict__ A, const half_t* __restrict__ B,
            const float* __restrict__ bias,
            float* __restrict__ Cf, half_t* __restrict__ Ch,
            int M, int N, int K) {
    const int BK = 32;
    __shared__ half_t As[128][BK + 8];
    __shared__ half_t Bs[128][BK + 8];

    int bm = blockIdx.x * 128;
    int bn = blockIdx.y * 128;
    int wave = threadIdx.x >> 5;
    int lane = threadIdx.x & 31;
    int ln = lane & 15, lh = lane >> 4;
    int wm = (wave & 3) * 32;   // wave M offset in block
    int wn = (wave >> 2) * 64;  // wave N offset in block

    v8f acc[2][4];
#pragma unroll
    for (int i = 0; i < 2; ++i)
#pragma unroll
        for (int j = 0; j < 4; ++j) acc[i][j] = (v8f)0.f;

    int trow = threadIdx.x >> 1;
    int tcol = (threadIdx.x & 1) * 16;

    for (int k0 = 0; k0 < K; k0 += BK) {
        const half_t* aglob = &A[(size_t)(bm + trow) * K + k0 + tcol];
        const half_t* bglob = &B[(size_t)(bn + trow) * K + k0 + tcol];
#if HAVE_ASYNC_LDS
        // CDNA5 async DMA into LDS, tracked by ASYNCcnt
        __builtin_amdgcn_global_load_async_to_lds_b128(
            (as1_v4i*)(uintptr_t)aglob,       (as3_v4i*)(uintptr_t)&As[trow][tcol],     0, 0);
        __builtin_amdgcn_global_load_async_to_lds_b128(
            (as1_v4i*)(uintptr_t)(aglob + 8), (as3_v4i*)(uintptr_t)&As[trow][tcol + 8], 0, 0);
        __builtin_amdgcn_global_load_async_to_lds_b128(
            (as1_v4i*)(uintptr_t)bglob,       (as3_v4i*)(uintptr_t)&Bs[trow][tcol],     0, 0);
        __builtin_amdgcn_global_load_async_to_lds_b128(
            (as1_v4i*)(uintptr_t)(bglob + 8), (as3_v4i*)(uintptr_t)&Bs[trow][tcol + 8], 0, 0);
        __builtin_amdgcn_s_wait_asynccnt(0);
#else
        {
            const uint4* asrc = (const uint4*)aglob;
            uint4* adst = (uint4*)&As[trow][tcol];
            adst[0] = asrc[0]; adst[1] = asrc[1];
            const uint4* bsrc = (const uint4*)bglob;
            uint4* bdst = (uint4*)&Bs[trow][tcol];
            bdst[0] = bsrc[0]; bdst[1] = bsrc[1];
        }
#endif
        if (k0 + BK < K) {  // hint next tiles into cache (global_prefetch_b8)
            __builtin_prefetch(&A[(size_t)(bm + trow) * K + k0 + BK + tcol], 0, 1);
            __builtin_prefetch(&B[(size_t)(bn + trow) * K + k0 + BK + tcol], 0, 1);
        }
        __syncthreads();

        v16h a0 = load_a_frag(&As[wm][0],      BK + 8);
        v16h a1 = load_a_frag(&As[wm + 16][0], BK + 8);
#pragma unroll
        for (int j = 0; j < 4; ++j) {
            v16h b = load_b_frag(&Bs[wn + j * 16][0], BK + 8);
            acc[0][j] = wmma16x16x32(a0, b, acc[0][j]);
            acc[1][j] = wmma16x16x32(a1, b, acc[1][j]);
        }
        __syncthreads();
    }

    // epilogue: bias add, store (D layout: lane half picks rows m+0..7 / m+8..15)
#pragma unroll
    for (int i = 0; i < 2; ++i) {
#pragma unroll
        for (int j = 0; j < 4; ++j) {
            int m0 = bm + wm + i * 16 + lh * 8;
            int n  = bn + wn + j * 16 + ln;
            float bv = bias[n];
#pragma unroll
            for (int r = 0; r < 8; ++r) {
                float val = acc[i][j][r] + bv;
                if (OUT_HALF) Ch[(size_t)(m0 + r) * N + n] = (half_t)val;
                else          Cf[(size_t)(m0 + r) * N + n] = val;
            }
        }
    }
}

// ---------------------------------------------------------------------------
// Kernel 4: split heads, RMSNorm(q,k) over head dim, RoPE, V transposed.
// grid = (L, H), 128 threads (one per head-dim element).
// ---------------------------------------------------------------------------
__global__ void qk_norm_rope(const half_t* __restrict__ qkv,  // [L][3*H*DH]
                             const float* __restrict__ pe,    // [1][1][L][64][2][2]
                             const float* __restrict__ q_scale,
                             const float* __restrict__ k_scale,
                             half_t* __restrict__ qh,   // [H][L][DH]
                             half_t* __restrict__ kh,   // [H][L][DH]
                             half_t* __restrict__ vT) { // [H][DH][L]
    int l = blockIdx.x, h = blockIdx.y, d = threadIdx.x;
    size_t row = (size_t)l * (3 * H_NUM * DH);
    float qv = (float)qkv[row + (0 * H_NUM + h) * DH + d];
    float kv = (float)qkv[row + (1 * H_NUM + h) * DH + d];
    float vv = (float)qkv[row + (2 * H_NUM + h) * DH + d];

    // sum of squares over 128 threads (4 waves)
    float sq = qv * qv, sk = kv * kv;
#pragma unroll
    for (int m = 1; m < 32; m <<= 1) {
        sq += __shfl_xor(sq, m, 32);
        sk += __shfl_xor(sk, m, 32);
    }
    __shared__ float pq[4], pk[4];
    int wid = threadIdx.x >> 5;
    if ((threadIdx.x & 31) == 0) { pq[wid] = sq; pk[wid] = sk; }
    __syncthreads();
    float rq = rsqrtf((pq[0] + pq[1] + pq[2] + pq[3]) * (1.f / DH) + 1e-6f);
    float rk = rsqrtf((pk[0] + pk[1] + pk[2] + pk[3]) * (1.f / DH) + 1e-6f);

    __shared__ float qs[DH], ks[DH];
    qs[d] = qv * rq * q_scale[d];
    ks[d] = kv * rk * k_scale[d];
    __syncthreads();

    // RoPE: out[2j+s] = pe[l][j][s][0]*x[2j] + pe[l][j][s][1]*x[2j+1]
    int j = d >> 1, s = d & 1;
    const float* peb = pe + (((size_t)l * (DH / 2) + j) * 2 + s) * 2;
    float qr = peb[0] * qs[2 * j] + peb[1] * qs[2 * j + 1];
    float kr = peb[0] * ks[2 * j] + peb[1] * ks[2 * j + 1];

    qh[((size_t)h * SEQLEN + l) * DH + d] = (half_t)qr;
    kh[((size_t)h * SEQLEN + l) * DH + d] = (half_t)kr;
    vT[((size_t)h * DH + d) * SEQLEN + l] = (half_t)vv;
}

// ---------------------------------------------------------------------------
// Kernel 5: flash attention. grid = (L/128, H), 256 threads = 8 waves.
// Each wave owns 16 query rows; loops over key chunks of 128 with online
// softmax; row stats live in a 16-lane shuffle group (wave32 D layout).
// launch_bounds(256,1): keep Q frags + S + O accumulators resident (no spills).
// ---------------------------------------------------------------------------
__global__ void __launch_bounds__(256, 1)
flash_attn(const half_t* __restrict__ qh, const half_t* __restrict__ kh,
           const half_t* __restrict__ vT, half_t* __restrict__ o) {
    const float scale = 0.08838834764831845f;  // 1/sqrt(128)
    int h = blockIdx.y;
    int q0 = blockIdx.x * 128;
    int wave = threadIdx.x >> 5;
    int lane = threadIdx.x & 31;
    int ln = lane & 15, lh = lane >> 4;

    const half_t* __restrict__ Q  = qh + (size_t)h * SEQLEN * DH;
    const half_t* __restrict__ Kt = kh + (size_t)h * SEQLEN * DH;
    const half_t* __restrict__ Vt = vT + (size_t)h * DH * SEQLEN;

    // Q fragments for this wave's 16 rows, full DH=128 (4 k-steps), kept live.
    v16h qa[4];
#pragma unroll
    for (int kk = 0; kk < 4; ++kk)
        qa[kk] = load_a_frag(Q + (size_t)(q0 + wave * 16) * DH + kk * 32, DH);

    v8f oacc[8];
#pragma unroll
    for (int j = 0; j < 8; ++j) oacc[j] = (v8f)0.f;
    float rowmax[8], rowsum[8];
#pragma unroll
    for (int r = 0; r < 8; ++r) { rowmax[r] = -1e30f; rowsum[r] = 0.f; }

    // Per-wave private P staging (wave reads only rows it wrote; DS ops are
    // in-order within a wave, so no block barrier is needed).
    __shared__ half_t Ps[128][128 + 8];

    for (int kc = 0; kc < SEQLEN; kc += 128) {
        // ---- S = scale * Q @ K^T for 16 rows x 128 key columns ----
        v8f s[8];
#pragma unroll
        for (int j = 0; j < 8; ++j) s[j] = (v8f)0.f;
#pragma unroll
        for (int kk = 0; kk < 4; ++kk) {
#pragma unroll
            for (int j = 0; j < 8; ++j) {
                v16h b = load_b_frag(Kt + (size_t)(kc + j * 16) * DH + kk * 32, DH);
                s[j] = wmma16x16x32(qa[kk], b, s[j]);
            }
        }
        // ---- online softmax row statistics ----
#pragma unroll
        for (int r = 0; r < 8; ++r) {
            float mloc = -1e30f;
#pragma unroll
            for (int j = 0; j < 8; ++j) {
                s[j][r] *= scale;
                mloc = fmaxf(mloc, s[j][r]);
            }
#pragma unroll
            for (int m = 1; m < 16; m <<= 1)
                mloc = fmaxf(mloc, __shfl_xor(mloc, m, 32));
            float mnew = fmaxf(rowmax[r], mloc);
            float alpha = __expf(rowmax[r] - mnew);
            rowmax[r] = mnew;
            rowsum[r] *= alpha;
#pragma unroll
            for (int j = 0; j < 8; ++j) oacc[j][r] *= alpha;
        }
        // ---- P = exp(S - m); stash in LDS for A-fragment reload ----
#pragma unroll
        for (int j = 0; j < 8; ++j) {
#pragma unroll
            for (int r = 0; r < 8; ++r) {
                float p = __expf(s[j][r] - rowmax[r]);
                rowsum[r] += p;  // per-lane partial; reduced at the end
                Ps[wave * 16 + lh * 8 + r][j * 16 + ln] = (half_t)p;
            }
        }
        // ---- O += P @ V (V pre-transposed: B[k=key][n=dh] = Vt[dh][key]) ----
#pragma unroll
        for (int kk = 0; kk < 4; ++kk) {
            v16h pa = load_a_frag(&Ps[wave * 16][kk * 32], 128 + 8);
#pragma unroll
            for (int j = 0; j < 8; ++j) {
                v16h vb = load_b_frag(Vt + (size_t)(j * 16) * SEQLEN + kc + kk * 32, SEQLEN);
                oacc[j] = wmma16x16x32(pa, vb, oacc[j]);
            }
        }
    }

    // finalize: reduce per-lane partial row sums across the 16-lane group
#pragma unroll
    for (int r = 0; r < 8; ++r) {
#pragma unroll
        for (int m = 1; m < 16; m <<= 1)
            rowsum[r] += __shfl_xor(rowsum[r], m, 32);
    }
    // store O as [l][h*DH + d], row-major stride DMODEL for the proj GEMM
#pragma unroll
    for (int j = 0; j < 8; ++j) {
#pragma unroll
        for (int r = 0; r < 8; ++r) {
            int l = q0 + wave * 16 + lh * 8 + r;
            int c = h * DH + j * 16 + ln;
            o[(size_t)l * DMODEL + c] = (half_t)(oacc[j][r] / rowsum[r]);
        }
    }
}

// ---------------------------------------------------------------------------
// Host-side orchestration
// ---------------------------------------------------------------------------
extern "C" void kernel_launch(void* const* d_in, const int* in_sizes, int n_in,
                              void* d_out, int out_size, void* d_ws, size_t ws_size,
                              hipStream_t stream) {
    const float* x         = (const float*)d_in[0];
    const float* pe        = (const float*)d_in[1];
    const float* qkv_w     = (const float*)d_in[2];
    const float* qkv_b     = (const float*)d_in[3];
    const float* qkv_down  = (const float*)d_in[4];
    const float* qkv_up    = (const float*)d_in[5];
    const float* proj_w    = (const float*)d_in[6];
    const float* proj_b    = (const float*)d_in[7];
    const float* proj_down = (const float*)d_in[8];
    const float* proj_up   = (const float*)d_in[9];
    const float* q_scale   = (const float*)d_in[10];
    const float* k_scale   = (const float*)d_in[11];
    float* out = (float*)d_out;

    const size_t L = SEQLEN, D = DMODEL;
    size_t off = 0;
    auto carve = [&](size_t bytes) {
        void* p = (char*)d_ws + off;
        off += (bytes + 255) & ~(size_t)255;
        return p;
    };
    half_t* xh  = (half_t*)carve(L * D * sizeof(half_t));               // x in f16
    half_t* w1  = (half_t*)carve(3 * D * D * sizeof(half_t));           // qkv weight (LoRA folded)
    half_t* w2  = (half_t*)carve(D * D * sizeof(half_t));               // proj weight (LoRA folded)
    half_t* qkv = (half_t*)carve(L * 3 * D * sizeof(half_t));           // qkv activations
    half_t* qh  = (half_t*)carve((size_t)H_NUM * L * DH * sizeof(half_t));
    half_t* kh  = (half_t*)carve((size_t)H_NUM * L * DH * sizeof(half_t));
    half_t* vT  = (half_t*)carve((size_t)H_NUM * DH * L * sizeof(half_t));
    half_t* oar = (half_t*)carve(L * D * sizeof(half_t));               // attn output
    (void)ws_size; (void)n_in; (void)in_sizes; (void)out_size;

    // 1) fold LoRA into weights (f16)
    {
        size_t n1 = 3 * D * D;
        fuse_lora_weight<<<(unsigned)((n1 + 255) / 256), 256, 0, stream>>>(
            qkv_w, qkv_up, qkv_down, w1, (int)(3 * D), (int)D);
        size_t n2 = D * D;
        fuse_lora_weight<<<(unsigned)((n2 + 255) / 256), 256, 0, stream>>>(
            proj_w, proj_up, proj_down, w2, (int)D, (int)D);
    }
    // 2) x -> f16
    cvt_f32_f16<<<(unsigned)((L * D + 255) / 256), 256, 0, stream>>>(x, xh, L * D);

    // 3) qkv = x @ W1^T + b  (M=2048, N=9216, K=3072), f16 out
    gemm_nt_f16<true><<<dim3(L / 128, (3 * D) / 128), 256, 0, stream>>>(
        xh, w1, qkv_b, nullptr, qkv, (int)L, (int)(3 * D), (int)D);

    // 4) head split + RMSNorm + RoPE + V transpose
    qk_norm_rope<<<dim3((unsigned)L, H_NUM), DH, 0, stream>>>(
        qkv, pe, q_scale, k_scale, qh, kh, vT);

    // 5) flash attention
    flash_attn<<<dim3(L / 128, H_NUM), 256, 0, stream>>>(qh, kh, vT, oar);

    // 6) out = o @ W2^T + proj_b  (M=2048, N=3072, K=3072), f32 out
    gemm_nt_f16<false><<<dim3(L / 128, D / 128), 256, 0, stream>>>(
        oar, w2, proj_b, out, nullptr, (int)L, (int)D, (int)D);
}
